// LanguageBrain_9423158248263
// MI455X (gfx1250) — matrix operand order repaired and checked by reading, not compiled, not run on hardware
//
#include <hip/hip_runtime.h>

// ---------------------------------------------------------------------------
// LanguageBrain on MI455X (gfx1250, wave32)
//
//  K1 embed+LN1      : bandwidth-bound gather+norm, writes X' = 0.7*LN1(x)
//  K2 W_out -> bf16^T: one-off prep so GEMM B-fragments are contiguous
//  K3 brain scan     : 32768 * 5 strictly serial 512x512 matvecs.
//                      Latency-bound -> single WGP (1 block, 512 thr = 16 waves),
//                      s in LDS (broadcast), h in registers, W L2-resident.
//  K4 LN2 -> bf16 Y  : aliased in-place over X' at 2KB row pitch
//  K5 head GEMM      : (32768x512)@(512x256)+b via v_wmma_f32_16x16x32_bf16,
//                      one 16x16 tile per wave, K-loop of 16 x 32.
// ---------------------------------------------------------------------------

#define D_MODEL   512
#define N_VOCAB   256
#define BATCH     32768
#define ROUNDS    5
#define LN_EPS    1e-5f

typedef __attribute__((ext_vector_type(16))) __bf16 v16bf;
typedef __attribute__((ext_vector_type(8)))  float  v8f;

union FragBF16 {
    uint4 u[2];
    v16bf v;
};

__device__ __forceinline__ unsigned short f32_to_bf16_rne(float f) {
    union { float f; unsigned int u; } cv; cv.f = f;
    unsigned int u = cv.u;
    unsigned int r = u + 0x7FFFu + ((u >> 16) & 1u);   // round-to-nearest-even
    return (unsigned short)(r >> 16);
}

// ---------------------------------------------------------------------------
// K1: embedding gather + LN1, scaled by 0.7 (folds the blend out of the scan).
// One row per block, 128 threads, float4 lanes, LDS tree reduction.
// ---------------------------------------------------------------------------
__global__ void embed_ln1_kernel(const int* __restrict__ tok,
                                 const float* __restrict__ emb,
                                 const float* __restrict__ g,
                                 const float* __restrict__ b,
                                 float* __restrict__ Xp) {
    const int row = blockIdx.x;
    const int t   = threadIdx.x;            // 0..127
    __shared__ float r1[128];
    __shared__ float r2[128];

    const int token = tok[row];
    const float4 v = *(const float4*)(emb + (size_t)token * D_MODEL + 4 * t);

    float s = v.x + v.y + v.z + v.w;
    float q = v.x * v.x + v.y * v.y + v.z * v.z + v.w * v.w;
    r1[t] = s; r2[t] = q;
    __syncthreads();
    #pragma unroll
    for (int off = 64; off > 0; off >>= 1) {
        if (t < off) { r1[t] += r1[t + off]; r2[t] += r2[t + off]; }
        __syncthreads();
    }
    const float mu   = r1[0] * (1.0f / D_MODEL);
    const float var  = r2[0] * (1.0f / D_MODEL) - mu * mu;
    const float rstd = rsqrtf(var + LN_EPS);

    float4 o;
    const int c0 = 4 * t;
    o.x = 0.7f * ((v.x - mu) * rstd * g[c0 + 0] + b[c0 + 0]);
    o.y = 0.7f * ((v.y - mu) * rstd * g[c0 + 1] + b[c0 + 1]);
    o.z = 0.7f * ((v.z - mu) * rstd * g[c0 + 2] + b[c0 + 2]);
    o.w = 0.7f * ((v.w - mu) * rstd * g[c0 + 3] + b[c0 + 3]);
    *(float4*)(Xp + (size_t)row * D_MODEL + c0) = o;
}

// ---------------------------------------------------------------------------
// K2: W_out (512x256 f32) -> W_outT (256x512 bf16). One-off, tiny.
// ---------------------------------------------------------------------------
__global__ void wout_prep_kernel(const float* __restrict__ Wout,
                                 unsigned short* __restrict__ WT) {
    const int idx = blockIdx.x * blockDim.x + threadIdx.x;   // 0..131071
    const int n = idx >> 9;          // 0..255
    const int k = idx & 511;         // 0..511
    WT[idx] = f32_to_bf16_rne(Wout[(size_t)k * N_VOCAB + n]);
}

// ---------------------------------------------------------------------------
// K3: serial brain scan. 1 block, 512 threads (16 waves, one WGP).
// Thread j owns output feature j. s broadcast from LDS; W column j streamed
// coalescedly from L2 (row-major row per k across the 512 threads).
// X' is consumed and overwritten in place with brain_out (same index, after).
// ---------------------------------------------------------------------------
__global__ void __launch_bounds__(512)
brain_scan_kernel(float* __restrict__ X,              // in: 0.7*LN1(x); out: brain_out
                  const float* __restrict__ W,        // 512x512 row-major
                  const float* __restrict__ bb,       // b_brain
                  const float* __restrict__ h0) {
    __shared__ float s_sh[D_MODEL];
    const int j = threadIdx.x;
    const float bj = bb[j];
    float h = h0[j];                                  // h is elementwise -> register

    for (int i = 0; i < BATCH; ++i) {
        float sv = X[(size_t)i * D_MODEL + j] + 0.3f * h;
        s_sh[j] = sv;
        __syncthreads();

        #pragma unroll 1
        for (int r = 0; r < ROUNDS; ++r) {
            float acc = bj;
            const float* wc = W + j;                  // column j, stride D_MODEL
            #pragma unroll 8
            for (int k = 0; k < D_MODEL; ++k) {
                acc = fmaf(s_sh[k], wc[(size_t)k << 9], acc);
            }
            sv = tanhf(acc);
            __syncthreads();                          // all lanes done reading s_sh
            if (r < ROUNDS - 1) {
                s_sh[j] = sv;
                __syncthreads();
            }
        }
        X[(size_t)i * D_MODEL + j] = sv;              // brain_out, in place
        h = 0.5f * sv + 0.5f * h;
    }
}

// ---------------------------------------------------------------------------
// K4: LN2 + f32->bf16, writing Y aliased into the same buffer at row pitch
// 2048 bytes (1024 bf16 slots, first 512 used). Reads finish before the first
// reduction barrier; writes happen after -> alias is safe.
// ---------------------------------------------------------------------------
__global__ void ln2_bf16_kernel(const float* __restrict__ BO,   // brain_out (= ws base)
                                const float* __restrict__ g,
                                const float* __restrict__ b,
                                unsigned short* __restrict__ Y) { // same ws base, pitch 1024
    const int row = blockIdx.x;
    const int t   = threadIdx.x;
    __shared__ float r1[128];
    __shared__ float r2[128];

    const float4 v = *(const float4*)(BO + (size_t)row * D_MODEL + 4 * t);
    float s = v.x + v.y + v.z + v.w;
    float q = v.x * v.x + v.y * v.y + v.z * v.z + v.w * v.w;
    r1[t] = s; r2[t] = q;
    __syncthreads();
    #pragma unroll
    for (int off = 64; off > 0; off >>= 1) {
        if (t < off) { r1[t] += r1[t + off]; r2[t] += r2[t + off]; }
        __syncthreads();
    }
    const float mu   = r1[0] * (1.0f / D_MODEL);
    const float var  = r2[0] * (1.0f / D_MODEL) - mu * mu;
    const float rstd = rsqrtf(var + LN_EPS);

    const int c0 = 4 * t;
    ushort4 o;
    o.x = f32_to_bf16_rne((v.x - mu) * rstd * g[c0 + 0] + b[c0 + 0]);
    o.y = f32_to_bf16_rne((v.y - mu) * rstd * g[c0 + 1] + b[c0 + 1]);
    o.z = f32_to_bf16_rne((v.z - mu) * rstd * g[c0 + 2] + b[c0 + 2]);
    o.w = f32_to_bf16_rne((v.w - mu) * rstd * g[c0 + 3] + b[c0 + 3]);
    *(ushort4*)(Y + (size_t)row * 1024 + c0) = o;     // pitch 1024 bf16 (aliased)
}

// ---------------------------------------------------------------------------
// K5: head GEMM via WMMA bf16 -> f32. Grid (2048, 4), block 128 (4 waves).
// Each wave owns one 16(M) x 16(N) tile; K loop: 16 steps of 32.
// Fragment layouts follow the CDNA5 ISA 16-bit A(16x32) / B(32x16) tables:
//   A: lane l -> row M=l&15, K-halves selected by l>>4 (8+8 contiguous bf16)
//   B: lane l -> col N=l&15, K-halves selected by l>>4 (8+8 contiguous bf16 of W_outT)
//   C/D: VGPR v, lane l -> M = v + 8*(l>>4), N = l&15
// ---------------------------------------------------------------------------
__global__ void __launch_bounds__(128)
head_gemm_kernel(const unsigned short* __restrict__ Y,   // pitch 1024 bf16 elems
                 const unsigned short* __restrict__ WT,  // 256 x 512 bf16
                 const float* __restrict__ bout,
                 float* __restrict__ out) {              // 32768 x 256 f32
    const int lane = threadIdx.x & 31;
    const int wave = threadIdx.x >> 5;
    const int row0 = blockIdx.x * 16;
    const int n0   = (blockIdx.y * 4 + wave) * 16;
    const int mloc = lane & 15;
    const int hi   = lane >> 4;

    const unsigned short* Arow = Y  + (size_t)(row0 + mloc) * 1024;
    const unsigned short* Brow = WT + (size_t)(n0  + mloc) * 512;

    v8f acc = {};
    #pragma unroll 4
    for (int kt = 0; kt < 16; ++kt) {
        const int k0 = kt * 32;
        FragBF16 a, bfrag;
        a.u[0] = *(const uint4*)(Arow + k0 + 8 * hi);
        a.u[1] = *(const uint4*)(Arow + k0 + 16 + 8 * hi);
        bfrag.u[0] = *(const uint4*)(Brow + k0 + 8 * hi);
        bfrag.u[1] = *(const uint4*)(Brow + k0 + 16 + 8 * hi);
        acc = __builtin_amdgcn_wmma_f32_16x16x32_bf16(
                  /*neg_a=*/false, a.v, /*neg_b=*/false, bfrag.v,
                  /*c_mod=*/(short)0, acc, /*reuse_a=*/false, /*reuse_b=*/false);
    }

    const float bias = bout[n0 + mloc];
    float* o = out + (size_t)row0 * N_VOCAB + n0 + mloc;
    #pragma unroll
    for (int v = 0; v < 8; ++v) {
        const int m = v + 8 * hi;
        o[(size_t)m * N_VOCAB] = acc[v] + bias;
    }
}

// ---------------------------------------------------------------------------
// Launch. Input order (setup_inputs): byte_input, emb_table, ln1_g, ln1_b,
// W_brain, b_brain, ln2_g, ln2_b, W_out, b_out, h0.
// Workspace: [0, 64MB)  X' f32 -> brain_out f32 -> Y bf16 (aliased, pitch 2KB)
//            [64MB, +256KB) W_outT bf16
// ---------------------------------------------------------------------------
extern "C" void kernel_launch(void* const* d_in, const int* in_sizes, int n_in,
                              void* d_out, int out_size, void* d_ws, size_t ws_size,
                              hipStream_t stream) {
    const int*   tok     = (const int*)  d_in[0];
    const float* emb     = (const float*)d_in[1];
    const float* ln1_g   = (const float*)d_in[2];
    const float* ln1_b   = (const float*)d_in[3];
    const float* W_brain = (const float*)d_in[4];
    const float* b_brain = (const float*)d_in[5];
    const float* ln2_g   = (const float*)d_in[6];
    const float* ln2_b   = (const float*)d_in[7];
    const float* W_out   = (const float*)d_in[8];
    const float* b_out   = (const float*)d_in[9];
    const float* h0      = (const float*)d_in[10];

    float*          Xp  = (float*)d_ws;
    unsigned short* Ybf = (unsigned short*)d_ws;   // aliased over Xp, pitch 1024
    unsigned short* WT  = (unsigned short*)((char*)d_ws + (size_t)BATCH * D_MODEL * 4);
    float*       logits = (float*)d_out;

    embed_ln1_kernel<<<BATCH, 128, 0, stream>>>(tok, emb, ln1_g, ln1_b, Xp);
    wout_prep_kernel<<<(D_MODEL * N_VOCAB) / 256, 256, 0, stream>>>(W_out, WT);
    brain_scan_kernel<<<1, D_MODEL, 0, stream>>>(Xp, W_brain, b_brain, h0);
    ln2_bf16_kernel<<<BATCH, 128, 0, stream>>>(Xp, ln2_g, ln2_b, Ybf);
    head_gemm_kernel<<<dim3(BATCH / 16, 4), 128, 0, stream>>>(Ybf, WT, b_out, logits);
}
